// Net_4922032521378
// MI455X (gfx1250) — matrix operand order repaired and checked by reading, not compiled, not run on hardware
//
#include <hip/hip_runtime.h>
#include <hip/hip_bf16.h>

// ---------------- constants from the reference ----------------
#define NUM_USERS 50000
#define NUM_ITEMS 50000
#define TOTAL_N   (NUM_USERS + NUM_ITEMS)
#define ED  64
#define NH  4
#define HE  (NH * ED)       // 256
#define BB  16384
#define E_U  300000
#define E_I  300000
#define E_UI 600000

typedef __bf16 v16bf __attribute__((ext_vector_type(16)));
typedef __bf16 v8bf  __attribute__((ext_vector_type(8)));
typedef float  v8f   __attribute__((ext_vector_type(8)));

#define GID     (blockIdx.x * (long)blockDim.x + threadIdx.x)
#define GSTRIDE (gridDim.x * (long)blockDim.x)

__device__ __forceinline__ __bf16 f2bf(float f) {
    unsigned u = __float_as_uint(f);
    u += 0x7FFFu + ((u >> 16) & 1u);          // round-to-nearest-even
    unsigned short h = (unsigned short)(u >> 16);
    __bf16 r;
    __builtin_memcpy(&r, &h, sizeof(r));
    return r;
}

__device__ __forceinline__ void atomicMaxF32(float* addr, float val) {
    unsigned* ua = (unsigned*)addr;
    unsigned old = *ua;
    while (true) {
        float f = __uint_as_float(old);
        if (f >= val) break;
        unsigned assumed = old;
        old = atomicCAS(ua, assumed, __float_as_uint(val));
        if (old == assumed) break;
    }
}

// ---------------------------------------------------------------------------
// WMMA bf16 GEMM:  C[M,N] = act( A[M,K] @ W[K,N] + bias )
//   - A: bf16 row-major (pre-converted), rows optionally gathered via rowIdx
//   - Wt: bf16 N x K row-major (= W transposed) -> B fragment is 2 b128 loads
//   - C (f32) and/or Cbf (bf16) outputs, each nullable
//   - one wave per 16x16 tile; M,N mult of 16; K mult of 32
//   Inner loop: 4x global_load_b128 + 1x v_wmma_f32_16x16x32_bf16
// ---------------------------------------------------------------------------
__global__ void k_wmma_gemm(const __bf16* __restrict__ A, int lda,
                            const int* __restrict__ rowIdx, int idxOff,
                            const __bf16* __restrict__ Wt,
                            const float* __restrict__ bias,
                            float* __restrict__ C, int ldc,
                            __bf16* __restrict__ Cbf, int ldcb,
                            int M, int N, int K, int act) {
    const int lane = threadIdx.x & 31;
    const int half = lane >> 4;        // 0: lanes 0-15, 1: lanes 16-31
    const int l15  = lane & 15;
    const int wave = (int)(GID >> 5);
    const int nw   = (int)(GSTRIDE >> 5);
    const int tilesN = N >> 4;
    const int tiles  = (M >> 4) * tilesN;

    for (int t = wave; t < tiles; t += nw) {
        const int tm = t / tilesN, tn = t % tilesN;
        const int m0 = tm << 4,   n0 = tn << 4;

        const int row  = m0 + l15;                           // A fragment row
        const long arow = rowIdx ? (long)(rowIdx[row] + idxOff) : (long)row;
        const __bf16* Ap = A + arow * (long)lda;
        const int ncol = n0 + l15;                           // B fragment column
        const __bf16* Bp = Wt + (long)ncol * K;

        v8f acc = {};
        for (int k0 = 0; k0 < K; k0 += 32) {
            // A 16x32: per lane two contiguous 8-elem runs (ISA 16-bit A layout)
            const v8bf a_lo = *(const v8bf*)(Ap + k0 + half * 8);
            const v8bf a_hi = *(const v8bf*)(Ap + k0 + 16 + half * 8);
            // B 32x16: per lane two contiguous 8-elem runs (same K order as A)
            const v8bf b_lo = *(const v8bf*)(Bp + k0 + half * 8);
            const v8bf b_hi = *(const v8bf*)(Bp + k0 + 16 + half * 8);
            const v16bf a = __builtin_shufflevector(a_lo, a_hi,
                0, 1, 2, 3, 4, 5, 6, 7, 8, 9, 10, 11, 12, 13, 14, 15);
            const v16bf b = __builtin_shufflevector(b_lo, b_hi,
                0, 1, 2, 3, 4, 5, 6, 7, 8, 9, 10, 11, 12, 13, 14, 15);
            acc = __builtin_amdgcn_wmma_f32_16x16x32_bf16(
                false, a, false, b, (short)0, acc, false, false);
        }

        const float bv = bias ? bias[ncol] : 0.0f;
#pragma unroll
        for (int j = 0; j < 8; ++j) {
            const int rm = m0 + j + (half ? 8 : 0);
            float v = acc[j] + bv;
            if (act == 1) v = fmaxf(v, 0.0f);
            if (C)   C[(long)rm * ldc + ncol]   = v;
            if (Cbf) Cbf[(long)rm * ldcb + ncol] = f2bf(v);
        }
    }
}

// --------------------- conversion / transpose kernels ---------------------

__global__ void k_f32_to_bf16(const float* __restrict__ in, __bf16* __restrict__ outp, long n) {
    for (long i = GID; i < n; i += GSTRIDE) outp[i] = f2bf(in[i]);
}

// W: K x N row-major f32  ->  Wt: N x K row-major bf16
__global__ void k_w_transpose(const float* __restrict__ W, __bf16* __restrict__ Wt, int K, int N) {
    const long total = (long)K * N;
    for (long i = GID; i < total; i += GSTRIDE) {
        const int k = (int)(i / N), n = (int)(i % N);
        Wt[(long)n * K + k] = f2bf(W[i]);
    }
}

// --------------------- GAT helper kernels ---------------------

// al_s[n,h] = <h[n,h,:], a_src[h,:]> ; al_d likewise
__global__ void k_attn_logits(const float* __restrict__ h,
                              const float* __restrict__ asrc,
                              const float* __restrict__ adst,
                              float* __restrict__ als, float* __restrict__ ald, int N) {
    const long total = (long)N * NH;
    for (long i = GID; i < total; i += GSTRIDE) {
        const int n = (int)(i >> 2), hh = (int)(i & 3);
        const float* hp = h + (long)n * HE + hh * ED;
        const float* as = asrc + hh * ED;
        const float* ad = adst + hh * ED;
        float ss = 0.f, sd = 0.f;
#pragma unroll 8
        for (int d = 0; d < ED; ++d) { const float hv = hp[d]; ss += hv * as[d]; sd += hv * ad[d]; }
        als[i] = ss; ald[i] = sd;
    }
}

__global__ void k_fill(float* __restrict__ p, float v, long n) {
    for (long i = GID; i < n; i += GSTRIDE) p[i] = v;
}

__global__ void k_init_bias(float* __restrict__ out, const float* __restrict__ bias, int N) {
    const long total = (long)N * HE;
    for (long i = GID; i < total; i += GSTRIDE) out[i] = bias[i & (HE - 1)];
}

// segment max of leaky_relu(al_s[src]+al_d[dst]) over dst ; self-loops appended
__global__ void k_edge_max(const float* __restrict__ als, const float* __restrict__ ald,
                           const int* __restrict__ src, const int* __restrict__ dst,
                           int E, int N, float* __restrict__ m) {
    const long total = (long)(E + N) * NH;
    for (long i = GID; i < total; i += GSTRIDE) {
        const int e = (int)(i >> 2), hh = (int)(i & 3);
        const int s = (e < E) ? src[e] : (e - E);
        const int d = (e < E) ? dst[e] : (e - E);
        float v = als[(long)s * NH + hh] + ald[(long)d * NH + hh];
        v = (v > 0.f) ? v : 0.2f * v;
        atomicMaxF32(&m[(long)d * NH + hh], v);
    }
}

// ex = exp(e - m[dst]); denom[dst] += ex
__global__ void k_edge_exp(const float* __restrict__ als, const float* __restrict__ ald,
                           const int* __restrict__ src, const int* __restrict__ dst,
                           int E, int N, const float* __restrict__ m,
                           float* __restrict__ ex, float* __restrict__ denom) {
    const long total = (long)(E + N) * NH;
    for (long i = GID; i < total; i += GSTRIDE) {
        const int e = (int)(i >> 2), hh = (int)(i & 3);
        const int s = (e < E) ? src[e] : (e - E);
        const int d = (e < E) ? dst[e] : (e - E);
        float v = als[(long)s * NH + hh] + ald[(long)d * NH + hh];
        v = (v > 0.f) ? v : 0.2f * v;
        const float e_ = __expf(v - m[(long)d * NH + hh]);
        ex[i] = e_;
        atomicAdd(&denom[(long)d * NH + hh], e_);
    }
}

// out[dst, c] += h[src, c] * ex[e,h]/denom[dst,h]
__global__ void k_edge_scatter(const float* __restrict__ h,
                               const int* __restrict__ src, const int* __restrict__ dst,
                               int E, int N, const float* __restrict__ ex,
                               const float* __restrict__ denom, float* __restrict__ out) {
    const long total = (long)(E + N) * HE;
    for (long i = GID; i < total; i += GSTRIDE) {
        const long e = i >> 8;
        const int  c = (int)(i & (HE - 1));
        const int  hh = c >> 6;
        const int s = (e < E) ? src[e] : (int)(e - E);
        const int d = (e < E) ? dst[e] : (int)(e - E);
        const float alpha = ex[e * NH + hh] / (denom[(long)d * NH + hh] + 1e-16f);
        atomicAdd(&out[(long)d * HE + c], h[(long)s * HE + c] * alpha);
    }
}

// --------------------- 2-token attention + head ---------------------
// one wave per batch row; writes combined features directly in bf16
__global__ void k_attn2(const float* __restrict__ Q, const float* __restrict__ K,
                        const float* __restrict__ V, __bf16* __restrict__ out,
                        int ldo, int off, int Brows) {
    const int lane = threadIdx.x & 31;
    const int wave = (int)(GID >> 5);
    const int nw   = (int)(GSTRIDE >> 5);
    for (int b = wave; b < Brows; b += nw) {
        const float* q0 = Q + (long)(2 * b) * ED; const float* q1 = q0 + ED;
        const float* k0 = K + (long)(2 * b) * ED; const float* k1 = k0 + ED;
        const float* v0 = V + (long)(2 * b) * ED; const float* v1 = v0 + ED;
        const int d = lane * 2;
        const float q0a = q0[d], q0b = q0[d + 1], q1a = q1[d], q1b = q1[d + 1];
        const float k0a = k0[d], k0b = k0[d + 1], k1a = k1[d], k1b = k1[d + 1];
        float s00 = q0a * k0a + q0b * k0b;
        float s01 = q0a * k1a + q0b * k1b;
        float s10 = q1a * k0a + q1b * k0b;
        float s11 = q1a * k1a + q1b * k1b;
        for (int o = 16; o > 0; o >>= 1) {
            s00 += __shfl_xor(s00, o, 32);
            s01 += __shfl_xor(s01, o, 32);
            s10 += __shfl_xor(s10, o, 32);
            s11 += __shfl_xor(s11, o, 32);
        }
        const float sc = 0.125f;                       // 1/sqrt(64)
        s00 *= sc; s01 *= sc; s10 *= sc; s11 *= sc;
        const float m0 = fmaxf(s00, s01), m1 = fmaxf(s10, s11);
        const float e00 = __expf(s00 - m0), e01 = __expf(s01 - m0);
        const float e10 = __expf(s10 - m1), e11 = __expf(s11 - m1);
        const float a00 = e00 / (e00 + e01), a01 = e01 / (e00 + e01);
        const float a10 = e10 / (e10 + e11), a11 = e11 / (e10 + e11);
        const float oa = 0.5f * ((a00 + a10) * v0[d]     + (a01 + a11) * v1[d]);
        const float ob = 0.5f * ((a00 + a10) * v0[d + 1] + (a01 + a11) * v1[d + 1]);
        out[(long)b * ldo + off + d]     = f2bf(oa);
        out[(long)b * ldo + off + d + 1] = f2bf(ob);
    }
}

// out[b] = sigmoid( <h1[b,:], w> + bias ) ; one wave per row
__global__ void k_fc2_sigmoid(const float* __restrict__ X, const float* __restrict__ w,
                              const float* __restrict__ bias, float* __restrict__ out, int Brows) {
    const int lane = threadIdx.x & 31;
    const int wave = (int)(GID >> 5);
    const int nw   = (int)(GSTRIDE >> 5);
    for (int b = wave; b < Brows; b += nw) {
        const float* x = X + (long)b * ED;
        float s = x[lane] * w[lane] + x[lane + 32] * w[lane + 32];
        for (int o = 16; o > 0; o >>= 1) s += __shfl_xor(s, o, 32);
        if (lane == 0) out[b] = 1.0f / (1.0f + __expf(-(s + bias[0])));
    }
}

// --------------------- host side ---------------------
static inline int eb(long total) {            // elementwise grid blocks (256 tpb)
    long b = (total + 255) / 256;
    if (b > 32768) b = 32768;
    if (b < 1) b = 1;
    return (int)b;
}

static inline void launch_gemm(const __bf16* A, int lda, const int* ridx, int roff,
                               const __bf16* Wt, const float* bias,
                               float* C, int ldc, __bf16* Cbf, int ldcb,
                               int M, int N, int K, int act, hipStream_t s) {
    int tiles = (M >> 4) * (N >> 4);
    int blocks = (tiles + 7) / 8;             // 8 waves / 256-thread block
    if (blocks > 16384) blocks = 16384;
    if (blocks < 1) blocks = 1;
    k_wmma_gemm<<<blocks, 256, 0, s>>>(A, lda, ridx, roff, Wt, bias, C, ldc, Cbf, ldcb,
                                       M, N, K, act);
}

extern "C" void kernel_launch(void* const* d_in, const int* in_sizes, int n_in,
                              void* d_out, int out_size, void* d_ws, size_t ws_size,
                              hipStream_t stream) {
    const int*   user_idx = (const int*)d_in[0];
    const int*   item_idx = (const int*)d_in[1];
    const int*   eiU  = (const int*)d_in[2];
    const int*   eiI  = (const int*)d_in[3];
    const int*   eiUI = (const int*)d_in[4];
    const float* emb  = (const float*)d_in[5];
    const float* WgA[3]   = {(const float*)d_in[6],  (const float*)d_in[10], (const float*)d_in[14]};
    const float* asrcA[3] = {(const float*)d_in[7],  (const float*)d_in[11], (const float*)d_in[15]};
    const float* adstA[3] = {(const float*)d_in[8],  (const float*)d_in[12], (const float*)d_in[16]};
    const float* bgA[3]   = {(const float*)d_in[9],  (const float*)d_in[13], (const float*)d_in[17]};
    const float* W_mu = (const float*)d_in[18]; const float* b_mu = (const float*)d_in[19];
    const float* W_mi = (const float*)d_in[20]; const float* b_mi = (const float*)d_in[21];
    const float* Wq_u = (const float*)d_in[22]; const float* bq_u = (const float*)d_in[23];
    const float* Wk_u = (const float*)d_in[24]; const float* bk_u = (const float*)d_in[25];
    const float* Wv_u = (const float*)d_in[26]; const float* bv_u = (const float*)d_in[27];
    const float* Wq_i = (const float*)d_in[28]; const float* bq_i = (const float*)d_in[29];
    const float* Wk_i = (const float*)d_in[30]; const float* bk_i = (const float*)d_in[31];
    const float* Wv_i = (const float*)d_in[32]; const float* bv_i = (const float*)d_in[33];
    const float* W_f1 = (const float*)d_in[34]; const float* b_f1 = (const float*)d_in[35];
    const float* W_f2 = (const float*)d_in[36]; const float* b_f2 = (const float*)d_in[37];
    (void)in_sizes; (void)n_in; (void)out_size; (void)ws_size;

    // ---- workspace carve-up (byte-based, 256B aligned chunks) ----
    char* wsb = (char*)d_ws;
    size_t woff = 0;
    auto carve = [&](size_t bytes) -> char* {
        char* p = wsb + woff;
        woff += (bytes + 255) & ~(size_t)255;
        return p;
    };
    float* outU   = (float*)carve(sizeof(float) * (size_t)NUM_USERS * HE);
    float* outI   = (float*)carve(sizeof(float) * (size_t)NUM_ITEMS * HE);
    float* outUI  = (float*)carve(sizeof(float) * (size_t)TOTAL_N * HE);
    float* hbuf   = (float*)carve(sizeof(float) * (size_t)TOTAL_N * HE);
    float* als    = (float*)carve(sizeof(float) * (size_t)TOTAL_N * NH);
    float* ald    = (float*)carve(sizeof(float) * (size_t)TOTAL_N * NH);
    float* mbuf   = (float*)carve(sizeof(float) * (size_t)TOTAL_N * NH);
    float* denom  = (float*)carve(sizeof(float) * (size_t)TOTAL_N * NH);
    float* exbuf  = (float*)carve(sizeof(float) * (size_t)(E_UI + TOTAL_N) * NH);
    float* Qb     = (float*)carve(sizeof(float) * (size_t)BB * 2 * ED);
    float* Kb     = (float*)carve(sizeof(float) * (size_t)BB * 2 * ED);
    float* Vb     = (float*)carve(sizeof(float) * (size_t)BB * 2 * ED);
    float* h1     = (float*)carve(sizeof(float) * (size_t)BB * ED);
    __bf16* emb_bf   = (__bf16*)carve(2 * (size_t)TOTAL_N * ED);
    __bf16* outU_bf  = (__bf16*)carve(2 * (size_t)NUM_USERS * HE);
    __bf16* outI_bf  = (__bf16*)carve(2 * (size_t)NUM_ITEMS * HE);
    __bf16* outUI_bf = (__bf16*)carve(2 * (size_t)TOTAL_N * HE);
    __bf16* seqU_bf  = (__bf16*)carve(2 * (size_t)BB * 2 * ED);
    __bf16* seqI_bf  = (__bf16*)carve(2 * (size_t)BB * 2 * ED);
    __bf16* comb_bf  = (__bf16*)carve(2 * (size_t)BB * 2 * ED);
    __bf16* WgT[3]; for (int g = 0; g < 3; ++g) WgT[g] = (__bf16*)carve(2 * (size_t)ED * HE);
    __bf16* WmuT  = (__bf16*)carve(2 * (size_t)HE * ED);
    __bf16* WmiT  = (__bf16*)carve(2 * (size_t)HE * ED);
    __bf16* WqTu  = (__bf16*)carve(2 * (size_t)ED * ED);
    __bf16* WkTu  = (__bf16*)carve(2 * (size_t)ED * ED);
    __bf16* WvTu  = (__bf16*)carve(2 * (size_t)ED * ED);
    __bf16* WqTi  = (__bf16*)carve(2 * (size_t)ED * ED);
    __bf16* WkTi  = (__bf16*)carve(2 * (size_t)ED * ED);
    __bf16* WvTi  = (__bf16*)carve(2 * (size_t)ED * ED);
    __bf16* Wf1T  = (__bf16*)carve(2 * (size_t)(2 * ED) * ED);

    // ---- one-time conversions (per call; deterministic) ----
    k_f32_to_bf16<<<eb((long)TOTAL_N * ED), 256, 0, stream>>>(emb, emb_bf, (long)TOTAL_N * ED);
    for (int g = 0; g < 3; ++g)
        k_w_transpose<<<eb((long)ED * HE), 256, 0, stream>>>(WgA[g], WgT[g], ED, HE);
    k_w_transpose<<<eb((long)HE * ED), 256, 0, stream>>>(W_mu, WmuT, HE, ED);
    k_w_transpose<<<eb((long)HE * ED), 256, 0, stream>>>(W_mi, WmiT, HE, ED);
    k_w_transpose<<<eb((long)ED * ED), 256, 0, stream>>>(Wq_u, WqTu, ED, ED);
    k_w_transpose<<<eb((long)ED * ED), 256, 0, stream>>>(Wk_u, WkTu, ED, ED);
    k_w_transpose<<<eb((long)ED * ED), 256, 0, stream>>>(Wv_u, WvTu, ED, ED);
    k_w_transpose<<<eb((long)ED * ED), 256, 0, stream>>>(Wq_i, WqTi, ED, ED);
    k_w_transpose<<<eb((long)ED * ED), 256, 0, stream>>>(Wk_i, WkTi, ED, ED);
    k_w_transpose<<<eb((long)ED * ED), 256, 0, stream>>>(Wv_i, WvTi, ED, ED);
    k_w_transpose<<<eb((long)(2 * ED) * ED), 256, 0, stream>>>(W_f1, Wf1T, 2 * ED, ED);

    // ---- three GAT convolutions (sequential, reuse hbuf/al/m/denom/ex) ----
    struct GDesc { const __bf16* X; const int* src; const int* dst; int N; int E;
                   const __bf16* Wt; const float* as; const float* ad; const float* bg;
                   float* out; __bf16* out_bf; };
    const GDesc G[3] = {
        { emb_bf,                          eiU,  eiU  + E_U,  NUM_USERS, E_U,  WgT[0], asrcA[0], adstA[0], bgA[0], outU,  outU_bf  },
        { emb_bf + (size_t)NUM_USERS * ED, eiI,  eiI  + E_I,  NUM_ITEMS, E_I,  WgT[1], asrcA[1], adstA[1], bgA[1], outI,  outI_bf  },
        { emb_bf,                          eiUI, eiUI + E_UI, TOTAL_N,   E_UI, WgT[2], asrcA[2], adstA[2], bgA[2], outUI, outUI_bf },
    };
    for (int g = 0; g < 3; ++g) {
        const GDesc& gd = G[g];
        // h = X @ Wg   (N x 64) @ (64 x 256), f32 out for attention math
        launch_gemm(gd.X, ED, nullptr, 0, gd.Wt, nullptr, hbuf, HE, nullptr, 0,
                    gd.N, HE, ED, 0, stream);
        k_attn_logits<<<eb((long)gd.N * NH), 256, 0, stream>>>(hbuf, gd.as, gd.ad, als, ald, gd.N);
        k_fill<<<eb((long)gd.N * NH), 256, 0, stream>>>(mbuf, -1e30f, (long)gd.N * NH);
        k_fill<<<eb((long)gd.N * NH), 256, 0, stream>>>(denom, 0.0f, (long)gd.N * NH);
        k_edge_max<<<eb((long)(gd.E + gd.N) * NH), 256, 0, stream>>>(als, ald, gd.src, gd.dst, gd.E, gd.N, mbuf);
        k_edge_exp<<<eb((long)(gd.E + gd.N) * NH), 256, 0, stream>>>(als, ald, gd.src, gd.dst, gd.E, gd.N, mbuf, exbuf, denom);
        k_init_bias<<<eb((long)gd.N * HE), 256, 0, stream>>>(gd.out, gd.bg, gd.N);
        k_edge_scatter<<<eb((long)(gd.E + gd.N) * HE), 256, 0, stream>>>(hbuf, gd.src, gd.dst, gd.E, gd.N, exbuf, denom, gd.out);
        // bf16 copy for the downstream gather-GEMMs
        k_f32_to_bf16<<<eb((long)gd.N * HE), 256, 0, stream>>>(gd.out, gd.out_bf, (long)gd.N * HE);
    }

    // ---- gathered projections into the 2-token sequences (B,2,64), bf16 out ----
    launch_gemm(outU_bf,  HE, user_idx, 0,         WmuT, b_mu, nullptr, 0, seqU_bf,      2 * ED, BB, ED, HE, 1, stream);
    launch_gemm(outI_bf,  HE, item_idx, 0,         WmiT, b_mi, nullptr, 0, seqI_bf,      2 * ED, BB, ED, HE, 1, stream);
    launch_gemm(outUI_bf, HE, user_idx, 0,         WmuT, b_mu, nullptr, 0, seqU_bf + ED, 2 * ED, BB, ED, HE, 1, stream);
    launch_gemm(outUI_bf, HE, item_idx, NUM_USERS, WmiT, b_mi, nullptr, 0, seqI_bf + ED, 2 * ED, BB, ED, HE, 1, stream);

    // ---- user tower attention ----
    launch_gemm(seqU_bf, ED, nullptr, 0, WqTu, bq_u, Qb, ED, nullptr, 0, 2 * BB, ED, ED, 0, stream);
    launch_gemm(seqU_bf, ED, nullptr, 0, WkTu, bk_u, Kb, ED, nullptr, 0, 2 * BB, ED, ED, 0, stream);
    launch_gemm(seqU_bf, ED, nullptr, 0, WvTu, bv_u, Vb, ED, nullptr, 0, 2 * BB, ED, ED, 0, stream);
    k_attn2<<<eb((long)BB * 32), 256, 0, stream>>>(Qb, Kb, Vb, comb_bf, 2 * ED, 0, BB);

    // ---- item tower attention (reuse Q/K/V buffers; stream-ordered) ----
    launch_gemm(seqI_bf, ED, nullptr, 0, WqTi, bq_i, Qb, ED, nullptr, 0, 2 * BB, ED, ED, 0, stream);
    launch_gemm(seqI_bf, ED, nullptr, 0, WkTi, bk_i, Kb, ED, nullptr, 0, 2 * BB, ED, ED, 0, stream);
    launch_gemm(seqI_bf, ED, nullptr, 0, WvTi, bv_i, Vb, ED, nullptr, 0, 2 * BB, ED, ED, 0, stream);
    k_attn2<<<eb((long)BB * 32), 256, 0, stream>>>(Qb, Kb, Vb, comb_bf, 2 * ED, ED, BB);

    // ---- head: h1 = relu(comb @ W_f1 + b_f1); out = sigmoid(h1 @ W_f2 + b_f2) ----
    launch_gemm(comb_bf, 2 * ED, nullptr, 0, Wf1T, b_f1, h1, ED, nullptr, 0, BB, ED, 2 * ED, 1, stream);
    k_fc2_sigmoid<<<eb((long)BB * 32), 256, 0, stream>>>(h1, W_f2, b_f2, (float*)d_out, BB);
}
